// MoELayer_63556926046600
// MI455X (gfx1250) — matrix operand order repaired and checked by reading, not compiled, not run on hardware
//
#include <hip/hip_runtime.h>
#include <hip/hip_bf16.h>

// ---------------- problem constants ----------------
#define S_   2048
#define D_   1024
#define H_   16
#define A_   64
#define EN_  8
#define ET_  2
#define DE_  512
#define DS_  2048
#define D3_  (3 * D_)

typedef __attribute__((ext_vector_type(16))) __bf16 bf16x16;
typedef __attribute__((ext_vector_type(8)))  __bf16 bf16x8;
typedef __attribute__((ext_vector_type(8)))  float  f32x8;
typedef int v4i __attribute__((vector_size(16)));  // matches async-LDS builtin proto

static __device__ inline f32x8 zero8() {
  f32x8 z = {0.f, 0.f, 0.f, 0.f, 0.f, 0.f, 0.f, 0.f};
  return z;
}

// A-fragment (16x32, 16-bit): lane holds row m = lane&15,
// K elems {ko..ko+7, ko+16..ko+23} with ko = (lane&16)?8:0.
static __device__ inline bf16x16 load_a_frag(const __bf16* __restrict__ p, int ld) {
  int lane = threadIdx.x & 31;
  const __bf16* q = p + (long)(lane & 15) * ld + ((lane & 16) ? 8 : 0);
  bf16x8 lo = *(const bf16x8*)q;
  bf16x8 hi = *(const bf16x8*)(q + 16);
  bf16x16 r;
#pragma unroll
  for (int i = 0; i < 8; ++i) { r[i] = lo[i]; r[i + 8] = hi[i]; }
  return r;
}

// B-fragment (32x16, 16-bit): lane holds column n = lane&15,
// K elems {ko..ko+7, ko+16..ko+23} with ko = (lane&16)?8:0 (matches A-fragment).
// Source is an N-major matrix slice: row n contiguous in K.
static __device__ inline bf16x16 load_b_frag(const __bf16* __restrict__ p, int ld) {
  int lane = threadIdx.x & 31;
  const __bf16* q = p + (long)(lane & 15) * ld + ((lane & 16) ? 8 : 0);
  bf16x8 lo = *(const bf16x8*)q;
  bf16x8 hi = *(const bf16x8*)(q + 16);
  bf16x16 r;
#pragma unroll
  for (int i = 0; i < 8; ++i) { r[i] = lo[i]; r[i + 8] = hi[i]; }
  return r;
}

static __device__ inline f32x8 wmma_bf16(bf16x16 a, bf16x16 b, f32x8 c) {
  return __builtin_amdgcn_wmma_f32_16x16x32_bf16(false, a, false, b, (short)0, c,
                                                 false, false);
}

// ---- async global->LDS staging (gfx1250 ASYNCcnt path, guarded) ----
static __device__ inline void async_copy16(const __bf16* g, __bf16* l) {
#if __has_builtin(__builtin_amdgcn_global_load_async_to_lds_b128)
  __builtin_amdgcn_global_load_async_to_lds_b128(
      (__attribute__((address_space(1))) v4i*)(v4i*)(__bf16*)g,
      (__attribute__((address_space(3))) v4i*)(v4i*)l, 0, 0);
#else
  *(bf16x8*)l = *(const bf16x8*)g;
#endif
}

static __device__ inline void wait_async_copies() {
#if __has_builtin(__builtin_amdgcn_s_wait_asynccnt)
  __builtin_amdgcn_s_wait_asynccnt(0);
#else
  asm volatile("s_wait_asynccnt 0x0" ::: "memory");
#endif
}

// ---------------- weight prep ----------------

// dst[n*K + k] = (bf16) src[k*N + n], batched.
__global__ void transpose_cvt_kernel(const float* __restrict__ src,
                                     __bf16* __restrict__ dst,
                                     int K, int N, long sstride, long dstride) {
  __shared__ float tile[32][33];
  const float* s = src + (long)blockIdx.z * sstride;
  __bf16* d = dst + (long)blockIdx.z * dstride;
  int nb = blockIdx.x * 32, kb = blockIdx.y * 32;
#pragma unroll
  for (int j = threadIdx.y; j < 32; j += 8) {
    int kk = kb + j, nn = nb + threadIdx.x;
    tile[j][threadIdx.x] = (kk < K && nn < N) ? s[(long)kk * N + nn] : 0.f;
  }
  __syncthreads();
#pragma unroll
  for (int j = threadIdx.y; j < 32; j += 8) {
    int nn = nb + j, kk = kb + threadIdx.x;
    if (nn < N && kk < K) d[(long)nn * K + kk] = (__bf16)tile[threadIdx.x][j];
  }
}

__global__ void cvt_kernel(const float* __restrict__ src, __bf16* __restrict__ dst,
                           long n) {
  long i = (long)blockIdx.x * blockDim.x + threadIdx.x;
  if (i < n) dst[i] = (__bf16)src[i];
}

// ---------------- rmsnorm ----------------
__global__ void rmsnorm_kernel(const float* __restrict__ x, const float* __restrict__ w,
                               __bf16* __restrict__ out_bf, float* __restrict__ out_f) {
  int row = blockIdx.x;
  const float* xr = x + (long)row * D_;
  __shared__ float red[256];
  float ss = 0.f;
  for (int i = threadIdx.x; i < D_; i += 256) { float v = xr[i]; ss += v * v; }
  red[threadIdx.x] = ss;
  __syncthreads();
  for (int s = 128; s > 0; s >>= 1) {
    if (threadIdx.x < s) red[threadIdx.x] += red[threadIdx.x + s];
    __syncthreads();
  }
  float r = rsqrtf(red[0] / (float)D_ + 1e-5f);
  for (int i = threadIdx.x; i < D_; i += 256) {
    float v = xr[i] * r * w[i];
    out_bf[(long)row * D_ + i] = (__bf16)v;
    if (out_f) out_f[(long)row * D_ + i] = v;
  }
}

// ------- blocked WMMA GEMM: C[M,N] = A[M,K] * Bt[N,K]^T (+ add) ----------------
// 128 threads = 4 waves; workgroup tile 64x64. B tile (64x32) staged in LDS via
// async global->LDS copies and shared by all 4 waves; each wave reuses its
// A-fragment across 4 N-tiles (4 WMMAs per A load).
__global__ __launch_bounds__(128) void gemm_kernel(const __bf16* __restrict__ Abf,
                                                   const __bf16* __restrict__ Bt,
                                                   float* __restrict__ C,
                                                   const float* __restrict__ add,
                                                   int M, int N, int K) {
  __shared__ __align__(32) __bf16 Bs[64][32];
  int w = threadIdx.x >> 5;
  int nb = blockIdx.x * 64, mb = blockIdx.y * 64;
  int mrow = mb + w * 16;

  int tn = threadIdx.x >> 1;         // 0..63 : B row (column of C)
  int tc = (threadIdx.x & 1) * 16;   // 0/16  : K sub-run

  f32x8 acc[4];
#pragma unroll
  for (int nt = 0; nt < 4; ++nt) acc[nt] = zero8();

  for (int k = 0; k < K; k += 32) {
    const __bf16* gsrc = Bt + (long)(nb + tn) * K + k + tc;
    async_copy16(gsrc, &Bs[tn][tc]);
    async_copy16(gsrc + 8, &Bs[tn][tc + 8]);
    wait_async_copies();
    __syncthreads();

    bf16x16 a = load_a_frag(Abf + (long)mrow * K + k, K);
#pragma unroll
    for (int nt = 0; nt < 4; ++nt)
      acc[nt] = wmma_bf16(a, load_b_frag(&Bs[nt * 16][0], 32), acc[nt]);
    __syncthreads();
  }

  int lane = threadIdx.x & 31;
  int col = lane & 15, rb = (lane & 16) ? 8 : 0;
#pragma unroll
  for (int nt = 0; nt < 4; ++nt) {
#pragma unroll
    for (int r = 0; r < 8; ++r) {
      long idx = (long)(mrow + rb + r) * N + nb + nt * 16 + col;
      float v = acc[nt][r];
      if (add) v += add[idx];
      C[idx] = v;
    }
  }
}

// ---------------- rotary + pack Q/K bf16 row-major, V transposed ----------------
__global__ void rope_pack_kernel(const float* __restrict__ qkv,
                                 __bf16* __restrict__ Qb, __bf16* __restrict__ Kb,
                                 __bf16* __restrict__ Vt) {
  int s = blockIdx.x, h = blockIdx.y, a = threadIdx.x;  // blockDim.x == 64
  const float* base = qkv + (long)s * D3_ + h * A_;
  int i = a & 31;
  float inv_freq = __expf(-(2.0f * (float)i / (float)A_) * 9.2103403719761836f);
  float ang = (float)s * inv_freq;
  float cf = __cosf(ang), sf = __sinf(ang);
  float q1 = base[i], q2 = base[i + 32];
  float k1 = base[D_ + i], k2 = base[D_ + i + 32];
  float qv = (a < 32) ? (q1 * cf + q2 * sf) : (-q1 * sf + q2 * cf);
  float kv = (a < 32) ? (k1 * cf + k2 * sf) : (-k1 * sf + k2 * cf);
  float vv = base[2 * D_ + a];
  long hs = (long)h * S_ + s;
  Qb[hs * A_ + a] = (__bf16)qv;
  Kb[hs * A_ + a] = (__bf16)kv;
  Vt[((long)h * A_ + a) * S_ + s] = (__bf16)vv;
}

// ---------------- flash attention: one wave per (head, 16-query tile) ----------
__global__ __launch_bounds__(32) void attn_kernel(const __bf16* __restrict__ Qb,
                                                  const __bf16* __restrict__ Kb,
                                                  const __bf16* __restrict__ Vt,
                                                  __bf16* __restrict__ Obf) {
  int qb = blockIdx.x, h = blockIdx.y;
  int qbase = qb * 16;
  int lane = threadIdx.x;
  int half = (lane & 16) ? 8 : 0;
  int l15 = lane & 15;

  const __bf16* Qh = Qb + ((long)h * S_ + qbase) * A_;
  const __bf16* Kh = Kb + (long)h * S_ * A_;
  const __bf16* Vh = Vt + (long)h * A_ * S_;

  bf16x16 q0 = load_a_frag(Qh, A_);        // head-dim K=0..31
  bf16x16 q1 = load_a_frag(Qh + 32, A_);   // head-dim K=32..63

  float m[8], lsum[8];
#pragma unroll
  for (int r = 0; r < 8; ++r) { m[r] = -3.0e38f; lsum[r] = 0.f; }
  f32x8 o0 = zero8(), o1 = zero8(), o2 = zero8(), o3 = zero8();

  __shared__ float Pl[16][36];

  int nkb = (qbase + 16 + 31) >> 5;
  for (int kb = 0; kb < nkb; ++kb) {
    int kbase = kb * 32;
    f32x8 s0 = zero8(), s1 = zero8();
    s0 = wmma_bf16(q0, load_b_frag(Kh + (long)kbase * A_, A_), s0);
    s0 = wmma_bf16(q1, load_b_frag(Kh + (long)kbase * A_ + 32, A_), s0);
    s1 = wmma_bf16(q0, load_b_frag(Kh + (long)(kbase + 16) * A_, A_), s1);
    s1 = wmma_bf16(q1, load_b_frag(Kh + (long)(kbase + 16) * A_ + 32, A_), s1);

    float alpha[8];
#pragma unroll
    for (int r = 0; r < 8; ++r) {
      int qrow = qbase + half + r;
      float v0 = s0[r] * 0.125f;       // 1/sqrt(64)
      float v1 = s1[r] * 0.125f;
      if (kbase + l15 > qrow)      v0 = -3.0e38f;  // causal mask
      if (kbase + 16 + l15 > qrow) v1 = -3.0e38f;
      float mx = fmaxf(v0, v1);
#pragma unroll
      for (int off = 8; off; off >>= 1) mx = fmaxf(mx, __shfl_xor(mx, off, 32));
      float nm = fmaxf(m[r], mx);
      float a = __expf(m[r] - nm);
      v0 = __expf(v0 - nm);
      v1 = __expf(v1 - nm);
      float rs = v0 + v1;
#pragma unroll
      for (int off = 8; off; off >>= 1) rs += __shfl_xor(rs, off, 32);
      lsum[r] = lsum[r] * a + rs;
      m[r] = nm;
      alpha[r] = a;
      Pl[half + r][l15] = v0;
      Pl[half + r][16 + l15] = v1;
    }
#pragma unroll
    for (int r = 0; r < 8; ++r) {
      o0[r] *= alpha[r]; o1[r] *= alpha[r]; o2[r] *= alpha[r]; o3[r] *= alpha[r];
    }
    __syncthreads();
    bf16x16 pf;  // P (16x32) as A-fragment, bf16
    {
      int ko = (lane & 16) ? 8 : 0;
#pragma unroll
      for (int i = 0; i < 8; ++i) {
        pf[i]     = (__bf16)Pl[l15][ko + i];
        pf[i + 8] = (__bf16)Pl[l15][ko + 16 + i];
      }
    }
    __syncthreads();
    o0 = wmma_bf16(pf, load_b_frag(Vh + (long)(0 * 16) * S_ + kbase, S_), o0);
    o1 = wmma_bf16(pf, load_b_frag(Vh + (long)(1 * 16) * S_ + kbase, S_), o1);
    o2 = wmma_bf16(pf, load_b_frag(Vh + (long)(2 * 16) * S_ + kbase, S_), o2);
    o3 = wmma_bf16(pf, load_b_frag(Vh + (long)(3 * 16) * S_ + kbase, S_), o3);
  }

#pragma unroll
  for (int r = 0; r < 8; ++r) {
    float inv = 1.0f / lsum[r];
    long row = (long)(qbase + half + r) * D_ + h * A_ + l15;
    Obf[row + 0]  = (__bf16)(o0[r] * inv);
    Obf[row + 16] = (__bf16)(o1[r] * inv);
    Obf[row + 32] = (__bf16)(o2[r] * inv);
    Obf[row + 48] = (__bf16)(o3[r] * inv);
  }
}

// ---------------- router ----------------
__global__ void router_logits_kernel(const float* __restrict__ xf,
                                     const float* __restrict__ Wr,
                                     float* __restrict__ logits) {
  int t = blockIdx.x;
  int e = threadIdx.x >> 5, lane = threadIdx.x & 31;
  const float* xr = xf + (long)t * D_;
  float s = 0.f;
  for (int d = lane; d < D_; d += 32) s += xr[d] * Wr[d * EN_ + e];
#pragma unroll
  for (int off = 16; off; off >>= 1) s += __shfl_xor(s, off, 32);
  if (lane == 0) logits[t * EN_ + e] = s;
}

__global__ void router_post_kernel(const float* __restrict__ logits,
                                   const float* __restrict__ bias,
                                   float* __restrict__ combine,
                                   float* __restrict__ stats) {
  int t = blockIdx.x * blockDim.x + threadIdx.x;
  if (t >= S_) return;
  float lg[EN_];
#pragma unroll
  for (int e = 0; e < EN_; ++e) lg[e] = logits[t * EN_ + e];
  int i0 = 0; float v0 = lg[0];
#pragma unroll
  for (int e = 1; e < EN_; ++e) if (lg[e] > v0) { v0 = lg[e]; i0 = e; }
  int i1 = (i0 == 0) ? 1 : 0; float v1 = lg[i1];
#pragma unroll
  for (int e = 0; e < EN_; ++e)
    if (e != i0 && lg[e] > v1) { v1 = lg[e]; i1 = e; }
  v0 += bias[i0]; v1 += bias[i1];
  float mx = fmaxf(v0, v1);
  float e0 = __expf(v0 - mx), e1 = __expf(v1 - mx);
  float inv = 1.0f / (e0 + e1);
#pragma unroll
  for (int e = 0; e < EN_; ++e) combine[t * EN_ + e] = 0.f;
  combine[t * EN_ + i0] = e0 * inv;      // ROUTED_SCALE == 1
  combine[t * EN_ + i1] += e1 * inv;
  atomicAdd(&stats[i0], 1.0f);
  atomicAdd(&stats[i1], 1.0f);
  float fm = lg[0];
#pragma unroll
  for (int e = 1; e < EN_; ++e) fm = fmaxf(fm, lg[e]);
  float sum = 0.f, p[EN_];
#pragma unroll
  for (int e = 0; e < EN_; ++e) { p[e] = __expf(lg[e] - fm); sum += p[e]; }
#pragma unroll
  for (int e = 0; e < EN_; ++e) atomicAdd(&stats[EN_ + e], p[e] / sum);
}

// ---- fused gated up-projection: h = silu(A*W1t^T)*(A*W3t^T)*scale -------------
// One wave per 16x32 strip: A-fragment reused across 4 WMMAs (g/u x 2 N-tiles).
__global__ __launch_bounds__(32) void gated_up_kernel(
    const __bf16* __restrict__ Abf, const __bf16* __restrict__ W1t,
    const __bf16* __restrict__ W3t, const float* __restrict__ scale,
    __bf16* __restrict__ Hout, int N, int K, long wstride, long hstride,
    int sstride) {
  int e = blockIdx.z;
  const __bf16* w1 = W1t + (long)e * wstride;
  const __bf16* w3 = W3t + (long)e * wstride;
  __bf16* hout = Hout + (long)e * hstride;
  int nb = blockIdx.x * 32, tb = blockIdx.y * 16;
  f32x8 g0 = zero8(), u0 = zero8(), g1 = zero8(), u1 = zero8();
  for (int k = 0; k < K; k += 32) {
    bf16x16 a = load_a_frag(Abf + (long)tb * K + k, K);
    g0 = wmma_bf16(a, load_b_frag(w1 + (long)nb * K + k, K), g0);
    u0 = wmma_bf16(a, load_b_frag(w3 + (long)nb * K + k, K), u0);
    g1 = wmma_bf16(a, load_b_frag(w1 + (long)(nb + 16) * K + k, K), g1);
    u1 = wmma_bf16(a, load_b_frag(w3 + (long)(nb + 16) * K + k, K), u1);
  }
  int lane = threadIdx.x & 31;
  int col = lane & 15, rb = (lane & 16) ? 8 : 0;
#pragma unroll
  for (int r = 0; r < 8; ++r) {
    int row = tb + rb + r;
    float sc = scale ? scale[(long)row * sstride + e] : 1.0f;
    float gv0 = g0[r], gv1 = g1[r];
    float h0 = (gv0 / (1.0f + __expf(-gv0))) * u0[r] * sc;
    float h1 = (gv1 / (1.0f + __expf(-gv1))) * u1[r] * sc;
    hout[(long)row * N + nb + col] = (__bf16)h0;
    hout[(long)row * N + nb + 16 + col] = (__bf16)h1;
  }
}

// ---- MoE down: Y[t,d] = sum_e sum_h hs[e,t,h] * W2[e,d,h] ---------------------
// One wave per 16x64 strip: A-fragment reused across 4 N-tiles.
__global__ __launch_bounds__(32) void moe_down_kernel(const __bf16* __restrict__ hs,
                                                      const __bf16* __restrict__ W2b,
                                                      float* __restrict__ Y) {
  int nb = blockIdx.x * 64, tb = blockIdx.y * 16;
  f32x8 acc[4];
#pragma unroll
  for (int nt = 0; nt < 4; ++nt) acc[nt] = zero8();
  for (int e = 0; e < EN_; ++e) {
    const __bf16* Ae = hs + (long)e * S_ * DE_;
    const __bf16* Be = W2b + (long)e * D_ * DE_;
    for (int k = 0; k < DE_; k += 32) {
      bf16x16 a = load_a_frag(Ae + (long)tb * DE_ + k, DE_);
#pragma unroll
      for (int nt = 0; nt < 4; ++nt)
        acc[nt] = wmma_bf16(a, load_b_frag(Be + (long)(nb + nt * 16) * DE_ + k, DE_),
                            acc[nt]);
    }
  }
  int lane = threadIdx.x & 31;
  int col = lane & 15, rb = (lane & 16) ? 8 : 0;
#pragma unroll
  for (int nt = 0; nt < 4; ++nt)
#pragma unroll
    for (int r = 0; r < 8; ++r)
      Y[(long)(tb + rb + r) * D_ + nb + nt * 16 + col] = acc[nt][r];
}

// ---------------- epilogue ----------------
__global__ void final_add_kernel(const float* __restrict__ a, const float* __restrict__ b,
                                 const float* __restrict__ c, float* __restrict__ out,
                                 long n) {
  long i = (long)blockIdx.x * blockDim.x + threadIdx.x;
  if (i < n) out[i] = a[i] + b[i] + c[i];
}

__global__ void loss_kernel(const float* __restrict__ stats, float* __restrict__ out) {
  float loss = 0.f;
#pragma unroll
  for (int e = 0; e < EN_; ++e) {
    float ind = stats[e] / (float)(S_ * ET_) * (float)EN_;
    float val = stats[EN_ + e] / (float)S_;
    loss += ind * val;
  }
  out[0] = loss;
}

// ---------------- host launcher ----------------
extern "C" void kernel_launch(void* const* d_in, const int* in_sizes, int n_in,
                              void* d_out, int out_size, void* d_ws, size_t ws_size,
                              hipStream_t stream) {
  (void)in_sizes; (void)n_in; (void)out_size; (void)ws_size;
  const float* x_input   = (const float*)d_in[0];
  const float* attn_w    = (const float*)d_in[1];
  const float* Wqkv      = (const float*)d_in[2];
  const float* Wo        = (const float*)d_in[3];
  const float* ffn_w     = (const float*)d_in[4];
  const float* keys_r    = (const float*)d_in[5];
  const float* keys_b    = (const float*)d_in[6];
  const float* experts   = (const float*)d_in[7];  // (3, EN, D, DE)
  const float* W_up      = (const float*)d_in[8];
  const float* W_down    = (const float*)d_in[9];
  float* out = (float*)d_out;

  char* p = (char*)d_ws;
  auto alloc = [&](size_t bytes) {
    char* r = p;
    p += (bytes + 255) & ~(size_t)255;
    return r;
  };
  __bf16* WqkvT   = (__bf16*)alloc((size_t)D_ * D3_ * 2);
  __bf16* WoT     = (__bf16*)alloc((size_t)D_ * D_ * 2);
  __bf16* E0T     = (__bf16*)alloc((size_t)EN_ * DE_ * D_ * 2);
  __bf16* E1T     = (__bf16*)alloc((size_t)EN_ * DE_ * D_ * 2);
  __bf16* E2b     = (__bf16*)alloc((size_t)EN_ * D_ * DE_ * 2);
  __bf16* WupT    = (__bf16*)alloc((size_t)2 * DS_ * D_ * 2);
  __bf16* WdnT    = (__bf16*)alloc((size_t)D_ * DS_ * 2);
  __bf16* xnorm   = (__bf16*)alloc((size_t)S_ * D_ * 2);
  float*  qkv     = (float*)alloc((size_t)S_ * D3_ * 4);
  __bf16* Qb      = (__bf16*)alloc((size_t)H_ * S_ * A_ * 2);
  __bf16* Kb      = (__bf16*)alloc((size_t)H_ * S_ * A_ * 2);
  __bf16* Vt      = (__bf16*)alloc((size_t)H_ * A_ * S_ * 2);
  __bf16* xattn   = (__bf16*)alloc((size_t)S_ * D_ * 2);
  float*  xffin   = (float*)alloc((size_t)S_ * D_ * 4);   // x_ffn_input
  float*  xffn_f  = (float*)alloc((size_t)S_ * D_ * 4);
  __bf16* xffn_bf = (__bf16*)alloc((size_t)S_ * D_ * 2);
  float*  logits  = (float*)alloc((size_t)S_ * EN_ * 4);
  float*  combine = (float*)alloc((size_t)S_ * EN_ * 4);
  float*  stats   = (float*)alloc(2 * EN_ * 4);            // counts | probsum
  __bf16* hs      = (__bf16*)alloc((size_t)EN_ * S_ * DE_ * 2);
  __bf16* hshared = (__bf16*)alloc((size_t)S_ * DS_ * 2);
  float*  ymoe    = (float*)alloc((size_t)S_ * D_ * 4);
  float*  yshared = (float*)alloc((size_t)S_ * D_ * 4);

  (void)hipMemsetAsync(stats, 0, 2 * EN_ * 4, stream);

  // ---- weight prep (bf16, N-major for B fragments) ----
  dim3 tb32(32, 8, 1);
  transpose_cvt_kernel<<<dim3(D3_ / 32, D_ / 32, 1), tb32, 0, stream>>>(
      Wqkv, WqkvT, D_, D3_, 0, 0);
  transpose_cvt_kernel<<<dim3(D_ / 32, D_ / 32, 1), tb32, 0, stream>>>(
      Wo, WoT, D_, D_, 0, 0);
  transpose_cvt_kernel<<<dim3(DE_ / 32, D_ / 32, EN_), tb32, 0, stream>>>(
      experts, E0T, D_, DE_, (long)D_ * DE_, (long)DE_ * D_);
  transpose_cvt_kernel<<<dim3(DE_ / 32, D_ / 32, EN_), tb32, 0, stream>>>(
      experts + (size_t)EN_ * D_ * DE_, E1T, D_, DE_, (long)D_ * DE_, (long)DE_ * D_);
  {
    long n = (long)EN_ * D_ * DE_;
    cvt_kernel<<<(unsigned)((n + 255) / 256), 256, 0, stream>>>(
        experts + (size_t)2 * EN_ * D_ * DE_, E2b, n);
  }
  transpose_cvt_kernel<<<dim3(2 * DS_ / 32, D_ / 32, 1), tb32, 0, stream>>>(
      W_up, WupT, D_, 2 * DS_, 0, 0);
  transpose_cvt_kernel<<<dim3(D_ / 32, DS_ / 32, 1), tb32, 0, stream>>>(
      W_down, WdnT, DS_, D_, 0, 0);

  // ---- attention path ----
  rmsnorm_kernel<<<S_, 256, 0, stream>>>(x_input, attn_w, xnorm, nullptr);
  gemm_kernel<<<dim3(D3_ / 64, S_ / 64, 1), 128, 0, stream>>>(
      xnorm, WqkvT, qkv, nullptr, S_, D3_, D_);
  rope_pack_kernel<<<dim3(S_, H_, 1), 64, 0, stream>>>(qkv, Qb, Kb, Vt);
  attn_kernel<<<dim3(S_ / 16, H_, 1), 32, 0, stream>>>(Qb, Kb, Vt, xattn);
  gemm_kernel<<<dim3(D_ / 64, S_ / 64, 1), 128, 0, stream>>>(
      xattn, WoT, xffin, x_input, S_, D_, D_);

  // ---- FFN path ----
  rmsnorm_kernel<<<S_, 256, 0, stream>>>(xffin, ffn_w, xffn_bf, xffn_f);
  router_logits_kernel<<<S_, 256, 0, stream>>>(xffn_f, keys_r, logits);
  router_post_kernel<<<S_ / 256, 256, 0, stream>>>(logits, keys_b, combine, stats);

  // MoE experts: fused gate/up + silu + combine-gate -> hs[e, t, h] (bf16)
  gated_up_kernel<<<dim3(DE_ / 32, S_ / 16, EN_), 32, 0, stream>>>(
      xffn_bf, E0T, E1T, combine, hs, DE_, D_, (long)DE_ * D_, (long)S_ * DE_, EN_);
  moe_down_kernel<<<dim3(D_ / 64, S_ / 16, 1), 32, 0, stream>>>(hs, E2b, ymoe);

  // shared FFN: fused gate/up (no scale), then down GEMM
  gated_up_kernel<<<dim3(DS_ / 32, S_ / 16, 1), 32, 0, stream>>>(
      xffn_bf, WupT, WupT + (size_t)DS_ * D_, nullptr, hshared, DS_, D_, 0, 0, 0);
  gemm_kernel<<<dim3(D_ / 64, S_ / 64, 1), 128, 0, stream>>>(
      hshared, WdnT, yshared, nullptr, S_, D_, DS_);

  // ---- epilogue ----
  {
    long n = (long)S_ * D_;
    final_add_kernel<<<(unsigned)((n + 255) / 256), 256, 0, stream>>>(
        ymoe, yshared, xffin, out, n);
  }
  loss_kernel<<<1, 1, 0, stream>>>(stats, out + (size_t)S_ * D_);
}
